// CausalLocalAttention_29119878267158
// MI455X (gfx1250) — matrix operand order repaired and checked by reading, not compiled, not run on hardware
//
#include <hip/hip_runtime.h>
#include <hip/hip_bf16.h>

// ---------------------------------------------------------------------------
// CDNA5 (gfx1250, wave32) windowed causal attention, bf16 WMMA everywhere.
// B=4, S=4096, D=1024, H=16, W=256, DH=64, NW=S/W=16.
// ---------------------------------------------------------------------------

typedef __attribute__((ext_vector_type(16))) __bf16 v16bf;
typedef __attribute__((ext_vector_type(8)))  float  v8f;

union FragBF {
    uint4  u[2];      // two 16-byte halves -> ds_load_b128 / global_load_b128
    v16bf  v;
};

__device__ __forceinline__ unsigned short f2bf(float f) {
    unsigned int u = __float_as_uint(f);
    unsigned int r = u + 0x7FFFu + ((u >> 16) & 1u);   // round-to-nearest-even
    return (unsigned short)(r >> 16);
}

// ---------------------------------------------------------------------------
// GEMM: Out[M,N] = A[M,K] * W[K,N] + bias[N]
//   A: f32 or bf16 (template), W: f32 (converted to bf16 on LDS stage)
//   Out: bf16 or f32 (template)
// Block: 256 threads (8 waves). Block tile 128x128, BK=32.
// Wave grid 2x4; each wave: 4x2 subtiles of 16x16 => 8 wmma per K step.
// Double-buffered LDS: next K-tile's global loads are issued before the
// current tile's WMMAs so VMEM latency overlaps matrix math; global_prefetch
// runs two tiles ahead.
// ---------------------------------------------------------------------------
template<bool A_IS_BF16, bool OUT_BF16>
__global__ __launch_bounds__(256)
void gemm_wmma_bf16(const void* __restrict__ Aptr,
                    const float* __restrict__ Wp,
                    const float* __restrict__ bias,
                    void* __restrict__ Out,
                    int M, int N, int K) {
    __shared__ __align__(16) unsigned short As[2][128 * 32];  // [m][k] row-major
    __shared__ __align__(16) unsigned short Bs[2][128 * 32];  // [n][k] (transposed)

    const int tid  = threadIdx.x;
    const int wid  = tid >> 5;
    const int lane = tid & 31;
    const int l16  = lane & 15;
    const int lh   = lane >> 4;            // 0 | 1 (half of wave)
    const int m_blk = blockIdx.y * 128;
    const int n_blk = blockIdx.x * 128;
    const int wm = wid >> 2;               // 0..1
    const int wn = wid & 3;                // 0..3

    // cooperative loader indices
    const int arow = tid >> 1;             // 0..127
    const int acol = (tid & 1) * 16;       // 0 | 16
    const int brow = tid >> 3;             // 0..31  (k within tile)
    const int bcol = (tid & 7) * 16;       // 0..112 (n within tile)

    v8f acc[4][2];
#pragma unroll
    for (int i = 0; i < 4; ++i)
#pragma unroll
        for (int j = 0; j < 2; ++j)
            acc[i][j] = (v8f){0.f, 0.f, 0.f, 0.f, 0.f, 0.f, 0.f, 0.f};

    // ---- global -> register loaders ----
    auto loadA = [&](int k0, uint4& d0, uint4& d1) {
        if (A_IS_BF16) {
            const unsigned short* Ap =
                (const unsigned short*)Aptr + (size_t)(m_blk + arow) * K + k0 + acol;
            d0 = *(const uint4*)Ap;
            d1 = *(const uint4*)(Ap + 8);
        } else {
            const float* Ap = (const float*)Aptr + (size_t)(m_blk + arow) * K + k0 + acol;
            unsigned short tmp[16];
#pragma unroll
            for (int i = 0; i < 4; ++i) {
                float4 f = *(const float4*)(Ap + i * 4);
                tmp[i * 4 + 0] = f2bf(f.x);
                tmp[i * 4 + 1] = f2bf(f.y);
                tmp[i * 4 + 2] = f2bf(f.z);
                tmp[i * 4 + 3] = f2bf(f.w);
            }
            d0 = *(const uint4*)&tmp[0];
            d1 = *(const uint4*)&tmp[8];
        }
    };
    auto loadB = [&](int k0, float4 (&f)[4]) {
        const float* Bp = Wp + (size_t)(k0 + brow) * N + n_blk + bcol;
#pragma unroll
        for (int i = 0; i < 4; ++i) f[i] = *(const float4*)(Bp + i * 4);
    };

    // ---- register -> LDS stagers ----
    auto stageA = [&](int buf, uint4 d0, uint4 d1) {
        *(uint4*)&As[buf][arow * 32 + acol]     = d0;
        *(uint4*)&As[buf][arow * 32 + acol + 8] = d1;
    };
    auto stageB = [&](int buf, const float4 (&f)[4]) {
#pragma unroll
        for (int i = 0; i < 4; ++i) {
            Bs[buf][(bcol + i * 4 + 0) * 32 + brow] = f2bf(f[i].x);
            Bs[buf][(bcol + i * 4 + 1) * 32 + brow] = f2bf(f[i].y);
            Bs[buf][(bcol + i * 4 + 2) * 32 + brow] = f2bf(f[i].z);
            Bs[buf][(bcol + i * 4 + 3) * 32 + brow] = f2bf(f[i].w);
        }
    };

    // ---- fragments + 8 WMMAs from one LDS buffer ----
    auto compute = [&](int buf) {
        FragBF a[4], b[2];
        const int kbA = lh * 8;   // A: lanes 0-15 -> K{0..7,16..23}; 16-31 -> K{8..15,24..31}
#pragma unroll
        for (int sm = 0; sm < 4; ++sm) {
            const unsigned short* p = &As[buf][(wm * 64 + sm * 16 + l16) * 32 + kbA];
            a[sm].u[0] = *(const uint4*)p;
            a[sm].u[1] = *(const uint4*)(p + 16);
        }
        const int kbB = lh * 8;   // B: lanes 0-15 -> K{0..7,16..23}; 16-31 -> K{8..15,24..31} (one column each)
#pragma unroll
        for (int sn = 0; sn < 2; ++sn) {
            const unsigned short* p = &Bs[buf][(wn * 32 + sn * 16 + l16) * 32 + kbB];
            b[sn].u[0] = *(const uint4*)p;
            b[sn].u[1] = *(const uint4*)(p + 16);
        }
#pragma unroll
        for (int sm = 0; sm < 4; ++sm)
#pragma unroll
            for (int sn = 0; sn < 2; ++sn)
                acc[sm][sn] = __builtin_amdgcn_wmma_f32_16x16x32_bf16(
                    false, a[sm].v, false, b[sn].v, (short)0, acc[sm][sn], false, false);
    };

    const int NK = K / 32;

    // prologue: stage tile 0
    {
        uint4 a0, a1; float4 bf4[4];
        loadA(0, a0, a1);
        loadB(0, bf4);
        stageA(0, a0, a1);
        stageB(0, bf4);
    }
    __syncthreads();

    for (int it = 0; it < NK; ++it) {
        const int k0 = it * 32;
        uint4 na0, na1; float4 nbf[4];
        if (it + 1 < NK) {
            // issue next tile's global loads before computing (overlap with WMMA)
            loadA(k0 + 32, na0, na1);
            loadB(k0 + 32, nbf);
        }
        if (it + 2 < NK) {
            // pull tile it+2 toward GL2 (global_prefetch_b8)
            const size_t ae = (size_t)(m_blk + arow) * K + (k0 + 64) + acol;
            __builtin_prefetch(A_IS_BF16
                                   ? (const void*)((const unsigned short*)Aptr + ae)
                                   : (const void*)((const float*)Aptr + ae), 0, 0);
            __builtin_prefetch(Wp + (size_t)(k0 + 64 + brow) * N + n_blk + bcol, 0, 0);
        }
        compute(it & 1);
        if (it + 1 < NK) {
            stageA((it + 1) & 1, na0, na1);
            stageB((it + 1) & 1, nbf);
        }
        __syncthreads();
    }

    // ---- epilogue: bias + store (C layout: M = j + 8*lh, N = l16) ----
#pragma unroll
    for (int sm = 0; sm < 4; ++sm) {
#pragma unroll
        for (int sn = 0; sn < 2; ++sn) {
            const int col = n_blk + wn * 32 + sn * 16 + l16;
            const float bv = bias[col];
            const int rbase = m_blk + wm * 64 + sm * 16 + lh * 8;
#pragma unroll
            for (int j = 0; j < 8; ++j) {
                float val = acc[sm][sn][j] + bv;
                if (OUT_BF16)
                    ((unsigned short*)Out)[(size_t)(rbase + j) * N + col] = f2bf(val);
                else
                    ((float*)Out)[(size_t)(rbase + j) * N + col] = val;
            }
        }
    }
}

// ---------------------------------------------------------------------------
// Attention: one block per (b, h, window). 256 threads = 8 waves.
// Q,K,V are bf16 in (B*S, D) layout, head cols = h*64 .. h*64+63.
// Each wave processes bands {wid, wid+8}: 16 query rows each.
// Scores via wmma (Q x K^T), causal per-tile skip + diagonal mask,
// f32 softmax, P through LDS, P x V via wmma, O scattered to (b,s,d).
// ---------------------------------------------------------------------------
__global__ __launch_bounds__(256)
void attn_wmma_bf16(const unsigned short* __restrict__ q,
                    const unsigned short* __restrict__ k,
                    const unsigned short* __restrict__ v,
                    unsigned short* __restrict__ o) {
    __shared__ __align__(16) unsigned short Vt[64 * 272];     // [dh][key], padded
    __shared__ __align__(16) unsigned short Ps[8 * 16 * 256]; // per-wave 16x256 P

    const int blk = blockIdx.x;
    const int b  = blk >> 8;
    const int h  = (blk >> 4) & 15;
    const int nb = blk & 15;
    const size_t row0 = (size_t)b * 4096 + (size_t)nb * 256;  // first row of window
    const int col0 = h * 64;

    const int tid  = threadIdx.x;
    const int wid  = tid >> 5;
    const int lane = tid & 31;
    const int l16  = lane & 15;
    const int lh   = lane >> 4;

    // ---- stage V transposed: Vt[dh][key]; thread = one key row ----
    {
        const unsigned short* vp = v + (row0 + tid) * 1024 + col0;
        unsigned short tmp[64];
#pragma unroll
        for (int i = 0; i < 8; ++i)
            *(uint4*)&tmp[i * 8] = *(const uint4*)(vp + i * 8);
#pragma unroll
        for (int d = 0; d < 64; ++d)
            Vt[d * 272 + tid] = tmp[d];
    }
    __syncthreads();

    const float scale = 0.125f;   // 1/sqrt(64)

    for (int band = wid; band < 16; band += 8) {
        // Q A-fragments for the two K-steps (dh 0..31, 32..63)
        FragBF aq[2];
#pragma unroll
        for (int kk = 0; kk < 2; ++kk) {
            const unsigned short* qp =
                q + (row0 + band * 16 + l16) * 1024 + col0 + kk * 32 + lh * 8;
            aq[kk].u[0] = *(const uint4*)qp;
            aq[kk].u[1] = *(const uint4*)(qp + 16);
        }

        float sc[16][8];
#pragma unroll
        for (int kt = 0; kt < 16; ++kt)
#pragma unroll
            for (int j = 0; j < 8; ++j)
                sc[kt][j] = -1e30f;

#pragma unroll
        for (int kt = 0; kt < 16; ++kt) {
            if (kt <= band) {   // causal: only lower-triangular key tiles
                v8f s = (v8f){0.f, 0.f, 0.f, 0.f, 0.f, 0.f, 0.f, 0.f};
#pragma unroll
                for (int kk = 0; kk < 2; ++kk) {
                    const unsigned short* kp =
                        k + (row0 + kt * 16 + l16) * 1024 + col0 + kk * 32 + lh * 8;
                    FragBF bk;
                    bk.u[0] = *(const uint4*)kp;
                    bk.u[1] = *(const uint4*)(kp + 16);
                    s = __builtin_amdgcn_wmma_f32_16x16x32_bf16(
                        false, aq[kk].v, false, bk.v, (short)0, s, false, false);
                }
#pragma unroll
                for (int j = 0; j < 8; ++j) {
                    float val = s[j] * scale;
                    if (kt == band) {
                        const int Mq = j + lh * 8;     // query row within tile
                        if (l16 > Mq) val = -1e30f;    // key col > query row
                    }
                    sc[kt][j] = val;
                }
            }
        }

        // ---- row softmax (row r: lanes share via 16-lane xor groups) ----
#pragma unroll
        for (int j = 0; j < 8; ++j) {
            float m = -1e30f;
#pragma unroll
            for (int kt = 0; kt < 16; ++kt) m = fmaxf(m, sc[kt][j]);
            m = fmaxf(m, __shfl_xor(m, 1));
            m = fmaxf(m, __shfl_xor(m, 2));
            m = fmaxf(m, __shfl_xor(m, 4));
            m = fmaxf(m, __shfl_xor(m, 8));
            float ssum = 0.f;
#pragma unroll
            for (int kt = 0; kt < 16; ++kt) {
                float p = __expf(sc[kt][j] - m);
                sc[kt][j] = p;
                ssum += p;
            }
            ssum += __shfl_xor(ssum, 1);
            ssum += __shfl_xor(ssum, 2);
            ssum += __shfl_xor(ssum, 4);
            ssum += __shfl_xor(ssum, 8);
            const float inv = 1.0f / ssum;
#pragma unroll
            for (int kt = 0; kt < 16; ++kt) sc[kt][j] *= inv;
        }

        // ---- write P band to LDS (C layout -> memory row-major) ----
        unsigned short* pbase = &Ps[wid * 16 * 256];
#pragma unroll
        for (int kt = 0; kt < 16; ++kt)
#pragma unroll
            for (int j = 0; j < 8; ++j)
                pbase[(j + lh * 8) * 256 + kt * 16 + l16] = f2bf(sc[kt][j]);

        // ---- O = P x V : 8 K-steps x 4 dh tiles ----
        v8f oacc[4];
#pragma unroll
        for (int nt = 0; nt < 4; ++nt)
            oacc[nt] = (v8f){0.f, 0.f, 0.f, 0.f, 0.f, 0.f, 0.f, 0.f};

#pragma unroll
        for (int kk2 = 0; kk2 < 256; kk2 += 32) {
            FragBF ap;
            const unsigned short* pp = pbase + l16 * 256 + kk2 + lh * 8;
            ap.u[0] = *(const uint4*)pp;
            ap.u[1] = *(const uint4*)(pp + 16);
#pragma unroll
            for (int nt = 0; nt < 4; ++nt) {
                const unsigned short* vp2 = &Vt[(nt * 16 + l16) * 272 + kk2 + lh * 8];
                FragBF bv_;
                bv_.u[0] = *(const uint4*)vp2;
                bv_.u[1] = *(const uint4*)(vp2 + 16);
                oacc[nt] = __builtin_amdgcn_wmma_f32_16x16x32_bf16(
                    false, ap.v, false, bv_.v, (short)0, oacc[nt], false, false);
            }
        }

        // ---- scatter O directly into (b, s, d) order ----
#pragma unroll
        for (int nt = 0; nt < 4; ++nt) {
            const int colg = col0 + nt * 16 + l16;
            const size_t rg = row0 + band * 16 + lh * 8;
#pragma unroll
            for (int j = 0; j < 8; ++j)
                o[(rg + j) * 1024 + colg] = f2bf(oacc[nt][j]);
        }
    }
}

// ---------------------------------------------------------------------------
// Host launch
// ---------------------------------------------------------------------------
extern "C" void kernel_launch(void* const* d_in, const int* in_sizes, int n_in,
                              void* d_out, int out_size, void* d_ws, size_t ws_size,
                              hipStream_t stream) {
    (void)in_sizes; (void)n_in; (void)out_size; (void)ws_size;

    const float* x  = (const float*)d_in[0];
    const float* Wq = (const float*)d_in[1];
    const float* bq = (const float*)d_in[2];
    const float* Wk = (const float*)d_in[3];
    const float* bk = (const float*)d_in[4];
    const float* Wv = (const float*)d_in[5];
    const float* bv = (const float*)d_in[6];
    const float* Wo = (const float*)d_in[7];
    const float* bo = (const float*)d_in[8];

    const int M = 4 * 4096;   // B*S = 16384
    const int N = 1024;
    const int K = 1024;

    const size_t SZ = (size_t)M * N * sizeof(unsigned short);  // 32 MB per buffer
    unsigned short* qb = (unsigned short*)d_ws;
    unsigned short* kb = (unsigned short*)((char*)d_ws + SZ);
    unsigned short* vb = (unsigned short*)((char*)d_ws + 2 * SZ);
    unsigned short* ob = (unsigned short*)((char*)d_ws + 3 * SZ);

    dim3 gg(N / 128, M / 128);   // (8, 128)
    dim3 bb(256);

    // QKV projections (f32 in, bf16 out)
    gemm_wmma_bf16<false, true><<<gg, bb, 0, stream>>>(x, Wq, bq, qb, M, N, K);
    gemm_wmma_bf16<false, true><<<gg, bb, 0, stream>>>(x, Wk, bk, kb, M, N, K);
    gemm_wmma_bf16<false, true><<<gg, bb, 0, stream>>>(x, Wv, bv, vb, M, N, K);

    // Windowed causal attention: B*H*NW = 4*16*16 = 1024 blocks
    attn_wmma_bf16<<<dim3(1024), bb, 0, stream>>>(qb, kb, vb, ob);

    // Output projection (bf16 in, f32 out)
    gemm_wmma_bf16<true, false><<<gg, bb, 0, stream>>>(ob, Wo, bo, (float*)d_out, M, N, K);
}
